// RetrievalLoss_3367254360220
// MI455X (gfx1250) — matrix operand order repaired and checked by reading, not compiled, not run on hardware
//
#include <hip/hip_runtime.h>
#include <hip/hip_bf16.h>
#include <math.h>

// ---------------------------------------------------------------------------
// RetrievalLoss on MI455X (gfx1250, wave32, WMMA)
//   stage 1: normalize targets rows -> f16 matrix bn (4 MB, L2-resident)
//   stage 2: fused bn@bn^T + diag-knockout + row-argmax via
//            v_wmma_f32_16x16x32_f16. Each wave owns M=64 rows (4 A tiles
//            resident in VGPRs) and sweeps 1/4 of the columns; 2-tile
//            register ping-pong hides L2 latency. All wave-level control
//            values go through readfirstlane so the diagonal fix-up and
//            loop control are pure scalar branches (EXEC stays all-1s,
//            as WMMA requires).
//   stage 3: combine split-argmax partials + hinge terms, deterministic
//            tree reductions -> scalar
// ---------------------------------------------------------------------------

typedef __attribute__((ext_vector_type(16))) _Float16 v16h;
typedef __attribute__((ext_vector_type(8)))  _Float16 v8h;
typedef __attribute__((ext_vector_type(4)))  _Float16 v4h;
typedef __attribute__((ext_vector_type(8)))  float    v8f;

#define DDIM   128   // feature dimension (fixed by reference)
#define MT     4     // 16-row tiles per wave => M = 64 rows resident
#define CSPLIT 4     // column-sweep splits (parallelism)

union U16h { v16h v; v8h h[2]; };

// ---------------- Stage 1: row-normalize targets, emit f16 -----------------
__global__ void __launch_bounds__(256)
normalize_to_half(const float* __restrict__ t, _Float16* __restrict__ bn, int N) {
  int r = blockIdx.x * blockDim.x + threadIdx.x;
  if (r >= N) return;
  const float4* row = (const float4*)(t + (size_t)r * DDIM);
  float4 buf[DDIM / 4];
  float s = 0.f;
#pragma unroll
  for (int i = 0; i < DDIM / 4; ++i) {
    float4 v = row[i];
    buf[i] = v;
    s += v.x * v.x + v.y * v.y + v.z * v.z + v.w * v.w;
  }
  float inv = 1.0f / sqrtf(s);
  v4h* out = (v4h*)(bn + (size_t)r * DDIM);
#pragma unroll
  for (int i = 0; i < DDIM / 4; ++i) {
    v4h h;
    h.x = (_Float16)(buf[i].x * inv);
    h.y = (_Float16)(buf[i].y * inv);
    h.z = (_Float16)(buf[i].z * inv);
    h.w = (_Float16)(buf[i].w * inv);
    out[i] = h;
  }
}

// ------- Stage 2: hardest-negative mining, WMMA GEMM + fused argmax --------
// jt, stripBase are SGPR-uniform; ln16/hi are per-lane.
__device__ __forceinline__
void mine_tile(const U16h (&A)[MT][4], const v16h (&B)[4], int jt,
               int stripBase, int ln16, int hi,
               v8f (&best)[MT], int (&bidx)[MT][8]) {
  v8f acc[MT];
  v8f zero = {};
#pragma unroll
  for (int t2 = 0; t2 < MT; ++t2) acc[t2] = zero;
#pragma unroll
  for (int s = 0; s < 4; ++s)
#pragma unroll
    for (int t2 = 0; t2 < MT; ++t2)
      acc[t2] = __builtin_amdgcn_wmma_f32_16x16x32_f16(false, A[t2][s].v, false, B[s],
                                                       (short)0, acc[t2], false, false);
  // Exact diagonal knockout (sim - eye). jt/stripBase are SGPRs, so this is
  // a scalar branch taken for at most MT (=4) column tiles of the whole sweep; the mask
  // is rematerialized here instead of staying live across the loop.
  if (__builtin_expect((unsigned)(jt - stripBase) < (unsigned)MT, 0)) {
#pragma unroll
    for (int t2 = 0; t2 < MT; ++t2)
      if (jt == stripBase + t2) {
#pragma unroll
        for (int v = 0; v < 8; ++v)
          acc[t2][v] -= ((hi * 8 + v) == ln16) ? 1.0f : 0.0f;
      }
  }
  // Running argmax; strict > keeps the lowest column index per lane.
  const int col = jt * 16 + ln16;   // scalar*imm + lane -> one v_add
#pragma unroll
  for (int t2 = 0; t2 < MT; ++t2)
#pragma unroll
    for (int v = 0; v < 8; ++v) {
      float x = acc[t2][v];
      bool g = x > best[t2][v];
      best[t2][v] = g ? x : best[t2][v];
      bidx[t2][v] = g ? col : bidx[t2][v];
    }
}

__global__ void __launch_bounds__(256, 1)
mine_hardest(const _Float16* __restrict__ bn, float* __restrict__ bvOut,
             int* __restrict__ biOut, int N) {
  const int lane = threadIdx.x & 31;
  const int ln16 = lane & 15;
  const int hi   = lane >> 4;               // 0: lanes 0-15, 1: lanes 16-31

  // Wave index: readfirstlane makes every derived control value an SGPR, so
  // the compiler emits scalar branches (no EXEC manipulation around WMMA).
  const int w = __builtin_amdgcn_readfirstlane(
      blockIdx.x * ((int)blockDim.x >> 5) + ((int)threadIdx.x >> 5));
  const int strip = w >> 2;                 // 64-row strip
  const int split = w & (CSPLIT - 1);       // column split
  const int rowBase = strip * (16 * MT);
  const int tps = (N / 16) / CSPLIT;        // tiles per split
  const int lo  = split * tps;
  const int stripBase = strip * MT;         // first diagonal tile index

  // A fragments (contiguous K: lane half hi holds k = 16*hi .. 16*hi+15): resident for the whole sweep.
  U16h A[MT][4];
#pragma unroll
  for (int t2 = 0; t2 < MT; ++t2) {
    const _Float16* rp = bn + (size_t)(rowBase + t2 * 16) * DDIM + ln16 * DDIM;
#pragma unroll
    for (int s = 0; s < 4; ++s) {
      int k0 = s * 32 + hi * 16;
      A[t2][s].h[0] = *(const v8h*)(rp + k0);        // K = k0 .. k0+7
      A[t2][s].h[1] = *(const v8h*)(rp + k0 + 8);    // K = k0+8 .. k0+15
    }
  }

  v8f best[MT]; int bidx[MT][8];
#pragma unroll
  for (int t2 = 0; t2 < MT; ++t2)
#pragma unroll
    for (int v = 0; v < 8; ++v) { best[t2][v] = -1e30f; bidx[t2][v] = 0; }

  // B fragments (contiguous K, same order as A): lane n holds column n,
  // contiguous K range -> one 32-byte load per k-step.
  // Scalar base pointer + single per-lane offset => SADDR-form loads.
  // 2-tile register ping-pong; bn has a 16-row slack tail so the
  // one-past-the-end prefetch is always in-bounds.
  const _Float16* basep = bn + (size_t)lo * 16 * DDIM;       // uniform
  const int laneOff = ln16 * DDIM + hi * 16;                 // per-lane
  v16h B0[4], B1[4];
#pragma unroll
  for (int s = 0; s < 4; ++s) B0[s] = *(const v16h*)(basep + laneOff + s * 32);

  for (int jt = lo; jt < lo + tps; jt += 2) {
#pragma unroll
    for (int s = 0; s < 4; ++s)
      B1[s] = *(const v16h*)(basep + laneOff + 16 * DDIM + s * 32);
    mine_tile(A, B0, jt, stripBase, ln16, hi, best, bidx);
#pragma unroll
    for (int s = 0; s < 4; ++s)
      B0[s] = *(const v16h*)(basep + laneOff + 32 * DDIM + s * 32);
    mine_tile(A, B1, jt + 1, stripBase, ln16, hi, best, bidx);
    basep += 32 * DDIM;                                      // scalar add
  }

  // Cross-lane argmax within each 16-lane half (rows live per VGPR slot).
  // Tie-break: smaller column index wins (matches jnp.argmax first-hit).
#pragma unroll
  for (int off = 8; off >= 1; off >>= 1)
#pragma unroll
    for (int t2 = 0; t2 < MT; ++t2)
#pragma unroll
      for (int v = 0; v < 8; ++v) {
        float ov = __shfl_xor(best[t2][v], off, 32);
        int   oi = __shfl_xor(bidx[t2][v], off, 32);
        if (ov > best[t2][v] || (ov == best[t2][v] && oi < bidx[t2][v])) {
          best[t2][v] = ov; bidx[t2][v] = oi;
        }
      }

  if (ln16 == 0) {  // lanes 0 and 16 hold the reduced result for their rows
#pragma unroll
    for (int t2 = 0; t2 < MT; ++t2) {
      int mb = rowBase + t2 * 16 + hi * 8;
#pragma unroll
      for (int v = 0; v < 8; ++v) {
        bvOut[(size_t)split * N + mb + v] = best[t2][v];
        biOut[(size_t)split * N + mb + v] = bidx[t2][v];
      }
    }
  }
}

// --------- Stage 3a: combine split argmax + per-row hinge terms ------------
__global__ void __launch_bounds__(256)
loss_partial(const float* __restrict__ q, const float* __restrict__ t,
             const float* __restrict__ bv, const int* __restrict__ bi,
             float* __restrict__ partial, int N) {
  int r = blockIdx.x * blockDim.x + threadIdx.x;
  float contrib = 0.f;
  if (r < N) {
    // Combine the CSPLIT partial argmaxes. Splits cover ascending column
    // ranges, so ascending scan with strict > keeps the first max.
    float bestV = bv[r];
    int   bestI = bi[r];
#pragma unroll
    for (int s = 1; s < CSPLIT; ++s) {
      float v2 = bv[(size_t)s * N + r];
      int   i2 = bi[(size_t)s * N + r];
      if (v2 > bestV || (v2 == bestV && i2 < bestI)) { bestV = v2; bestI = i2; }
    }
    const float4* qp = (const float4*)(q + (size_t)r * DDIM);
    const float4* tp = (const float4*)(t + (size_t)r * DDIM);
    const float4* wp = (const float4*)(t + (size_t)bestI * DDIM);
    float qq = 0.f, tt = 0.f, ww = 0.f, qt = 0.f, qw = 0.f;
#pragma unroll
    for (int i = 0; i < DDIM / 4; ++i) {
      float4 a = qp[i], b = tp[i], wv = wp[i];
      qq += a.x * a.x + a.y * a.y + a.z * a.z + a.w * a.w;
      tt += b.x * b.x + b.y * b.y + b.z * b.z + b.w * b.w;
      ww += wv.x * wv.x + wv.y * wv.y + wv.z * wv.z + wv.w * wv.w;
      qt += a.x * b.x + a.y * b.y + a.z * b.z + a.w * b.w;
      qw += a.x * wv.x + a.y * wv.y + a.z * wv.z + a.w * wv.w;
    }
    float pos = qt / (sqrtf(qq) * sqrtf(tt));
    float neg = qw / (sqrtf(qq) * sqrtf(ww));
    contrib = fmaxf(0.f, 1.0f - pos + neg);   // DELTA = 1.0
  }
  __shared__ float sm[256];
  sm[threadIdx.x] = contrib;
  __syncthreads();
  for (int s = 128; s > 0; s >>= 1) {
    if ((int)threadIdx.x < s) sm[threadIdx.x] += sm[threadIdx.x + s];
    __syncthreads();
  }
  if (threadIdx.x == 0) partial[blockIdx.x] = sm[0];
}

// --------------- Stage 3b: deterministic final reduction -------------------
__global__ void __launch_bounds__(256)
finalize(const float* __restrict__ partial, int nPart, float* __restrict__ out,
         float invN) {
  __shared__ float sm[256];
  float s = 0.f;
  for (int i = threadIdx.x; i < nPart; i += blockDim.x) s += partial[i];
  sm[threadIdx.x] = s;
  __syncthreads();
  for (int k = 128; k > 0; k >>= 1) {
    if ((int)threadIdx.x < k) sm[threadIdx.x] += sm[threadIdx.x + k];
    __syncthreads();
  }
  if (threadIdx.x == 0) out[0] = sm[0] * invN;
}

// ---------------------------------------------------------------------------
extern "C" void kernel_launch(void* const* d_in, const int* in_sizes, int n_in,
                              void* d_out, int out_size, void* d_ws, size_t ws_size,
                              hipStream_t stream) {
  const float* q = (const float*)d_in[0];   // queries [N, 128] f32
  const float* t = (const float*)d_in[1];   // targets [N, 128] f32
  const int N = in_sizes[1] / DDIM;         // 16384

  // Workspace layout (~4.75 MB):
  //   bn      : (N+16) * 128 f16   (16-row slack tail for pipeline prefetch)
  //   bestVal : CSPLIT * N  f32    (per-split row argmax value)
  //   bestIdx : CSPLIT * N  i32    (per-split row argmax index)
  //   partial : per-block sums
  char* ws = (char*)d_ws;
  _Float16* bn = (_Float16*)ws;
  size_t off = (size_t)(N + 16) * DDIM * sizeof(_Float16);
  float* bv = (float*)(ws + off);            off += (size_t)CSPLIT * N * sizeof(float);
  int*   bi = (int*)(ws + off);              off += (size_t)CSPLIT * N * sizeof(int);
  float* partial = (float*)(ws + off);

  const int nb = (N + 255) / 256;
  normalize_to_half<<<nb, 256, 0, stream>>>(t, bn, N);

  // One wave per (64-row strip, column split); 8 waves per block.
  const int waves  = (N / (16 * MT)) * CSPLIT;   // 1024
  const int blocks = waves / 8;                  // 128
  mine_hardest<<<blocks, 256, 0, stream>>>(bn, bv, bi, N);

  loss_partial<<<nb, 256, 0, stream>>>(q, t, bv, bi, partial, N);
  finalize<<<1, 256, 0, stream>>>(partial, nb, (float*)d_out, 1.0f / (float)N);
}
